// MultiheadAttention_59038620451474
// MI455X (gfx1250) — matrix operand order repaired and checked by reading, not compiled, not run on hardware
//
#include <hip/hip_runtime.h>
#include <hip/hip_bf16.h>
#include <math.h>

// ---------------- problem constants ----------------
constexpr int Bc  = 2;
constexpr int Lc  = 2048;
constexpr int Vc  = 1024;
constexpr int Hc  = 16;
constexpr int HDc = 64;        // head dim
constexpr float EPSc = 1e-3f;

typedef __attribute__((ext_vector_type(16))) _Float16 v16h;
typedef __attribute__((ext_vector_type(8)))  _Float16 v8h;
typedef __attribute__((ext_vector_type(8)))  float    v8f;

__device__ __forceinline__ int lane_id() { return (int)(threadIdx.x & 31); }

// ---------------- WMMA fragment helpers (V_WMMA_F32_16X16X32_F16 layouts) ----------------
// A fragment: 16x32 (MxK). lanes 0-15: row M=lane, elems 0..7 = K0..7, 8..15 = K16..23
//             lanes 16-31: row M=lane-16, elems 0..7 = K8..15, 8..15 = K24..31
__device__ __forceinline__ v16h load_a_f16(const _Float16* __restrict__ base, int ld,
                                           int row0, int k0) {
  int lane = lane_id();
  int r    = lane & 15;
  int koff = (lane & 16) ? 8 : 0;
  const _Float16* p = base + (size_t)(row0 + r) * ld + k0 + koff;
  v8h lo = *(const v8h*)(p);
  v8h hi = *(const v8h*)(p + 16);
  v16h a;
#pragma unroll
  for (int i = 0; i < 8; ++i) { a[i] = lo[i]; a[i + 8] = hi[i]; }
  return a;
}

// B fragment: 32x16 (KxN). lanes 0-15: col N=lane, elems 0..7 = K0..7, 8..15 = K16..23;
// lanes 16-31: col N=lane-16, elems 0..7 = K8..15, 8..15 = K24..31. Source stored column-major
// for B (row n of `base` holds B column n): B[k][n] = base[(col0+n)*ld + k0+k] -> two 16B loads.
__device__ __forceinline__ v16h load_bT_f16(const _Float16* __restrict__ base, int ld,
                                            int col0, int k0) {
  int lane = lane_id();
  int n    = lane & 15;
  int koff = (lane & 16) ? 8 : 0;
  const _Float16* p = base + (size_t)(col0 + n) * ld + k0 + koff;
  v8h lo = *(const v8h*)(p);
  v8h hi = *(const v8h*)(p + 16);
  v16h b;
#pragma unroll
  for (int i = 0; i < 8; ++i) { b[i] = lo[i]; b[i + 8] = hi[i]; }
  return b;
}

// C/D: VGPR r -> (M=r, N=lane) for lanes 0-15; (M=8+r, N=lane-16) for lanes 16-31.
__device__ __forceinline__ void store_c_f32(float* __restrict__ base, int ld,
                                            int row0, int col0, v8f c, float scale) {
  int lane = lane_id();
  int n    = col0 + (lane & 15);
  int m0   = row0 + ((lane & 16) ? 8 : 0);
#pragma unroll
  for (int r = 0; r < 8; ++r) base[(size_t)(m0 + r) * ld + n] = c[r] * scale;
}

__device__ __forceinline__ void store_c_f16(_Float16* __restrict__ base, int ld,
                                            int row0, int col0, v8f c) {
  int lane = lane_id();
  int n    = col0 + (lane & 15);
  int m0   = row0 + ((lane & 16) ? 8 : 0);
#pragma unroll
  for (int r = 0; r < 8; ++r) base[(size_t)(m0 + r) * ld + n] = (_Float16)c[r];
}

// Transposed f16 store: writes D^T, so row n of `base` holds output column n.
__device__ __forceinline__ void store_c_f16T(_Float16* __restrict__ base, int ld,
                                             int row0, int col0, v8f c) {
  int lane = lane_id();
  int n    = col0 + (lane & 15);
  int m0   = row0 + ((lane & 16) ? 8 : 0);
  v8h h;
#pragma unroll
  for (int r = 0; r < 8; ++r) h[r] = (_Float16)c[r];
  *(v8h*)(base + (size_t)n * ld + m0) = h;
}

#define WMMA_F16(a, b, c) \
  __builtin_amdgcn_wmma_f32_16x16x32_f16(false, (a), false, (b), (short)0, (c), false, false)

// ---------------- kernel 0a: fp32 -> f16 bulk convert of q/k/v activations ----------------
__global__ __launch_bounds__(256) void cvt_kernel(
    const float* __restrict__ q, const float* __restrict__ kt, const float* __restrict__ v,
    _Float16* __restrict__ oq, _Float16* __restrict__ ok, _Float16* __restrict__ ov) {
  int m = (int)blockIdx.y;
  const float* s = (m == 0 ? q : (m == 1 ? kt : v));
  _Float16*    d = (m == 0 ? oq : (m == 1 ? ok : ov));
  size_t i = ((size_t)blockIdx.x * 256 + threadIdx.x) * 8;
  float4 a = *(const float4*)(s + i);
  float4 b = *(const float4*)(s + i + 4);
  v8h h;
  h[0]=(_Float16)a.x; h[1]=(_Float16)a.y; h[2]=(_Float16)a.z; h[3]=(_Float16)a.w;
  h[4]=(_Float16)b.x; h[5]=(_Float16)b.y; h[6]=(_Float16)b.z; h[7]=(_Float16)b.w;
  *(v8h*)(d + i) = h;
}

// ---------------- kernel 0b: weight transpose+convert: WT[h][d][v] = (f16)W[h][v][d] ------
__global__ __launch_bounds__(256) void wt_kernel(
    const float* __restrict__ Wq, const float* __restrict__ Wk, const float* __restrict__ Wv,
    _Float16* __restrict__ Tq, _Float16* __restrict__ Tk, _Float16* __restrict__ Tv) {
  int m = (int)blockIdx.y;
  const float* W = (m == 0 ? Wq : (m == 1 ? Wk : Wv));
  _Float16*    T = (m == 0 ? Tq : (m == 1 ? Tk : Tv));
  size_t o = (size_t)blockIdx.x * 256 + threadIdx.x;   // index over H*HD*V
  int vv = (int)(o % Vc);
  int hd = (int)(o / Vc);
  int dd = hd % HDc;
  int hh = hd / HDc;
  T[o] = (_Float16)W[((size_t)hh * Vc + vv) * HDc + dd];
}

// ---------------- kernel 1: QKV projections (ping-pong pipelined, no reg rotation) --------
// grid = (Lc/64, 3*B*H), block = 128; wave w owns tileM = blockIdx.x*4+w, all 4 N-tiles.
// Q,K stored row-major (L,HD); V stored transposed (HD,L).
__global__ __launch_bounds__(128) void proj_kernel(
    const _Float16* __restrict__ Xq, const _Float16* __restrict__ Xk,
    const _Float16* __restrict__ Xv,
    const _Float16* __restrict__ Tq, const _Float16* __restrict__ Tk,
    const _Float16* __restrict__ Tv,
    _Float16* __restrict__ Qf, _Float16* __restrict__ Kf, _Float16* __restrict__ VfT) {
  int wave  = (int)(threadIdx.x >> 5);
  int tileM = (int)blockIdx.x * 4 + wave;   // 0..127
  int z     = (int)blockIdx.y;              // 0..95
  int mat   = z / (Bc * Hc);
  int bh    = z % (Bc * Hc);
  int b     = bh / Hc;
  int h     = bh % Hc;

  const _Float16* X  = (mat == 0 ? Xq : (mat == 1 ? Xk : Xv)) + (size_t)b * Lc * Vc;
  const _Float16* WT = (mat == 0 ? Tq : (mat == 1 ? Tk : Tv)) + (size_t)h * HDc * Vc;

  int row0 = tileM * 16;
  v8f acc[4] = {};

  // stage-0 prologue (k = 0)
  v16h a0 = load_a_f16(X, Vc, row0, 0);
  v16h b0[4];
#pragma unroll
  for (int nt = 0; nt < 4; ++nt) b0[nt] = load_bT_f16(WT, Vc, nt * 16, 0);

  // two k-steps (64 deep) per iteration; buffers alternate by NAME -> no v_mov rotation
  for (int k = 0; k < Vc - 64; k += 64) {
    v16h a1 = load_a_f16(X, Vc, row0, k + 32);
    v16h b1[4];
#pragma unroll
    for (int nt = 0; nt < 4; ++nt) b1[nt] = load_bT_f16(WT, Vc, nt * 16, k + 32);
#pragma unroll
    for (int nt = 0; nt < 4; ++nt) acc[nt] = WMMA_F16(a0, b0[nt], acc[nt]);

    a0 = load_a_f16(X, Vc, row0, k + 64);
#pragma unroll
    for (int nt = 0; nt < 4; ++nt) b0[nt] = load_bT_f16(WT, Vc, nt * 16, k + 64);
#pragma unroll
    for (int nt = 0; nt < 4; ++nt) acc[nt] = WMMA_F16(a1, b1[nt], acc[nt]);
  }
  // tail: steps at k = Vc-64 (in a0/b0) and k = Vc-32
  {
    v16h a1 = load_a_f16(X, Vc, row0, Vc - 32);
    v16h b1[4];
#pragma unroll
    for (int nt = 0; nt < 4; ++nt) b1[nt] = load_bT_f16(WT, Vc, nt * 16, Vc - 32);
#pragma unroll
    for (int nt = 0; nt < 4; ++nt) acc[nt] = WMMA_F16(a0, b0[nt], acc[nt]);
#pragma unroll
    for (int nt = 0; nt < 4; ++nt) acc[nt] = WMMA_F16(a1, b1[nt], acc[nt]);
  }

  if (mat == 2) {
    _Float16* O = VfT + (size_t)bh * HDc * Lc;
#pragma unroll
    for (int nt = 0; nt < 4; ++nt) store_c_f16T(O, Lc, row0, nt * 16, acc[nt]);
  } else {
    _Float16* O = (mat == 0 ? Qf : Kf) + (size_t)bh * Lc * HDc;
#pragma unroll
    for (int nt = 0; nt < 4; ++nt) store_c_f16(O, HDc, row0, nt * 16, acc[nt]);
  }
}

// ---------------- kernel 2: scores = Q K^T / sqrt(HD) for one (b,h) slab ----------------
// grid = (Lc/16, 8), block = 128; wave w owns 4 consecutive N-tiles.
// K-depth is only 64 -> load ALL fragments up front, then 8 back-to-back WMMAs.
__global__ __launch_bounds__(128) void scores_kernel(
    const _Float16* __restrict__ Qf, const _Float16* __restrict__ Kf,
    float* __restrict__ S) {
  int wave  = (int)(threadIdx.x >> 5);
  int tileM = (int)blockIdx.x;
  int tn0   = ((int)blockIdx.y * 4 + wave) * 4;   // 0,4,...,124

  v16h a0 = load_a_f16(Qf, HDc, tileM * 16, 0);
  v16h a1 = load_a_f16(Qf, HDc, tileM * 16, 32);
  v16h b0[4], b1[4];
#pragma unroll
  for (int nt = 0; nt < 4; ++nt) {
    b0[nt] = load_bT_f16(Kf, HDc, (tn0 + nt) * 16, 0);
    b1[nt] = load_bT_f16(Kf, HDc, (tn0 + nt) * 16, 32);
  }

  v8f acc[4] = {};
#pragma unroll
  for (int nt = 0; nt < 4; ++nt) acc[nt] = WMMA_F16(a0, b0[nt], acc[nt]);
#pragma unroll
  for (int nt = 0; nt < 4; ++nt) acc[nt] = WMMA_F16(a1, b1[nt], acc[nt]);

#pragma unroll
  for (int nt = 0; nt < 4; ++nt)
    store_c_f32(S, Lc, tileM * 16, (tn0 + nt) * 16, acc[nt], 0.125f);  // 1/sqrt(64)
}

// ---------------- kernel 3: full-row softmax + post-softmax masks -> f16 attn ------------
// Reference: softmax over the FULL row, then multiply by tril, qm[l], vm[m].
__global__ __launch_bounds__(256) void softmax_kernel(
    const float* __restrict__ S, _Float16* __restrict__ A,
    const int* __restrict__ qmask, const int* __restrict__ vmask) {
  __shared__ float red[256];
  int l = (int)blockIdx.x;
  int t = (int)threadIdx.x;
  const float* srow = S + (size_t)l * Lc;
  int m0 = t * 8;

  float4 a = *(const float4*)(srow + m0);
  float4 b = *(const float4*)(srow + m0 + 4);
  float x[8] = {a.x, a.y, a.z, a.w, b.x, b.y, b.z, b.w};

  float mx = x[0];
#pragma unroll
  for (int i = 1; i < 8; ++i) mx = fmaxf(mx, x[i]);
  red[t] = mx;
  __syncthreads();
  for (int s = 128; s > 0; s >>= 1) {
    if (t < s) red[t] = fmaxf(red[t], red[t + s]);
    __syncthreads();
  }
  mx = red[0];
  __syncthreads();

  float e[8];
  float sum = 0.0f;
#pragma unroll
  for (int i = 0; i < 8; ++i) { e[i] = __expf(x[i] - mx); sum += e[i]; }
  red[t] = sum;
  __syncthreads();
  for (int s = 128; s > 0; s >>= 1) {
    if (t < s) red[t] += red[t + s];
    __syncthreads();
  }
  float qs = (qmask[l] != 0) ? (1.0f / red[0]) : 0.0f;

  v8h h;
#pragma unroll
  for (int i = 0; i < 8; ++i) {
    int m = m0 + i;
    float p = ((m <= l) && (vmask[m] != 0)) ? e[i] * qs : 0.0f;
    h[i] = (_Float16)p;
  }
  *(v8h*)(A + (size_t)l * Lc + m0) = h;
}

// ---------------- kernel 4: heads = attn * V for one (b,h) slab (ping-pong pipelined) -----
// grid = Lc/64, block = 128; wave w owns tileM = blockIdx.x*4+w, all 4 N-tiles of HD.
// Step count clamped at causal diagonal; branches are wave-uniform (EXEC stays all-ones).
__global__ __launch_bounds__(128) void av_kernel(
    const _Float16* __restrict__ A, const _Float16* __restrict__ VfT,
    float* __restrict__ heads /* pre-offset: + b*L*V + h*HD, row stride V */) {
  int wave  = (int)(threadIdx.x >> 5);
  int tileM = (int)blockIdx.x * 4 + wave;
  int row0  = tileM * 16;
  int kend  = (tileM + 1) * 16;
  int steps = (kend + 31) >> 5;            // number of 32-deep k-steps (>= 1)
  v8f acc[4] = {};

  // stage-0 prologue (step 0 always valid; <=16 zeroed attn cols may be overread: harmless)
  v16h a0 = load_a_f16(A, Lc, row0, 0);
  v16h b0[4];
#pragma unroll
  for (int nt = 0; nt < 4; ++nt) b0[nt] = load_bT_f16(VfT, Lc, nt * 16, 0);

  int s = 0;
  for (; s + 2 < steps; s += 2) {
    v16h a1 = load_a_f16(A, Lc, row0, (s + 1) * 32);
    v16h b1[4];
#pragma unroll
    for (int nt = 0; nt < 4; ++nt) b1[nt] = load_bT_f16(VfT, Lc, nt * 16, (s + 1) * 32);
#pragma unroll
    for (int nt = 0; nt < 4; ++nt) acc[nt] = WMMA_F16(a0, b0[nt], acc[nt]);

    a0 = load_a_f16(A, Lc, row0, (s + 2) * 32);
#pragma unroll
    for (int nt = 0; nt < 4; ++nt) b0[nt] = load_bT_f16(VfT, Lc, nt * 16, (s + 2) * 32);
#pragma unroll
    for (int nt = 0; nt < 4; ++nt) acc[nt] = WMMA_F16(a1, b1[nt], acc[nt]);
  }
  if (s + 1 < steps) {                      // two steps remain: s, s+1
    v16h a1 = load_a_f16(A, Lc, row0, (s + 1) * 32);
    v16h b1[4];
#pragma unroll
    for (int nt = 0; nt < 4; ++nt) b1[nt] = load_bT_f16(VfT, Lc, nt * 16, (s + 1) * 32);
#pragma unroll
    for (int nt = 0; nt < 4; ++nt) acc[nt] = WMMA_F16(a0, b0[nt], acc[nt]);
#pragma unroll
    for (int nt = 0; nt < 4; ++nt) acc[nt] = WMMA_F16(a1, b1[nt], acc[nt]);
  } else {                                  // one step remains
#pragma unroll
    for (int nt = 0; nt < 4; ++nt) acc[nt] = WMMA_F16(a0, b0[nt], acc[nt]);
  }

#pragma unroll
  for (int nt = 0; nt < 4; ++nt)
    store_c_f32(heads, Vc, row0, nt * 16, acc[nt], 1.0f);
}

// ---------------- kernel 5: residual + LayerNorm ----------------
__global__ __launch_bounds__(256) void ln_kernel(
    const float* __restrict__ heads, const float* __restrict__ query,
    const float* __restrict__ gamma, const float* __restrict__ beta,
    float* __restrict__ out) {
  __shared__ float red[256];
  int row = (int)blockIdx.x;             // 0..B*L-1
  int t   = (int)threadIdx.x;
  int i0  = t * 4;
  const float* hr = heads + (size_t)row * Vc;
  const float* qr = query + (size_t)row * Vc;
  float*       orow = out + (size_t)row * Vc;

  float4 hv = *(const float4*)(hr + i0);
  float4 qv = *(const float4*)(qr + i0);
  float y[4] = {hv.x + qv.x, hv.y + qv.y, hv.z + qv.z, hv.w + qv.w};

  float s = y[0] + y[1] + y[2] + y[3];
  red[t] = s;
  __syncthreads();
  for (int st = 128; st > 0; st >>= 1) {
    if (t < st) red[t] += red[t + st];
    __syncthreads();
  }
  float mu = red[0] * (1.0f / Vc);
  __syncthreads();

  float vs = 0.0f;
#pragma unroll
  for (int i = 0; i < 4; ++i) { float d = y[i] - mu; vs += d * d; }
  red[t] = vs;
  __syncthreads();
  for (int st = 128; st > 0; st >>= 1) {
    if (t < st) red[t] += red[t + st];
    __syncthreads();
  }
  float rstd = rsqrtf(red[0] * (1.0f / Vc) + EPSc);

  float4 g  = *(const float4*)(gamma + i0);
  float4 be = *(const float4*)(beta + i0);
  float4 o;
  o.x = g.x * (y[0] - mu) * rstd + be.x;
  o.y = g.y * (y[1] - mu) * rstd + be.y;
  o.z = g.z * (y[2] - mu) * rstd + be.z;
  o.w = g.w * (y[3] - mu) * rstd + be.w;
  *(float4*)(orow + i0) = o;
}

// ---------------- host launcher ----------------
extern "C" void kernel_launch(void* const* d_in, const int* in_sizes, int n_in,
                              void* d_out, int out_size, void* d_ws, size_t ws_size,
                              hipStream_t stream) {
  const float* query = (const float*)d_in[0];
  const float* key_t = (const float*)d_in[1];
  const float* value = (const float*)d_in[2];
  const float* Wq    = (const float*)d_in[3];
  const float* Wk    = (const float*)d_in[4];
  const float* Wv    = (const float*)d_in[5];
  const float* gamma = (const float*)d_in[6];
  const float* beta  = (const float*)d_in[7];
  const int*   qmask = (const int*)d_in[8];
  const int*   vmask = (const int*)d_in[9];
  float* out = (float*)d_out;

  // ---- workspace layout (70 MB total) ----
  // [0,24MB): phase 1 = Xq/Xk/Xv f16 (8MB each); phase 2 = S f32 (16MB) + At f16 (8MB)
  char* ws = (char*)d_ws;
  const size_t MB = (size_t)1 << 20;
  _Float16* Xq  = (_Float16*)(ws);
  _Float16* Xk  = (_Float16*)(ws + 8 * MB);
  _Float16* Xv  = (_Float16*)(ws + 16 * MB);
  float*    S   = (float*)   (ws);            // aliases Xq/Xk (phase 2 only)
  _Float16* At  = (_Float16*)(ws + 16 * MB);  // aliases Xv   (phase 2 only)
  _Float16* Qf  = (_Float16*)(ws + 24 * MB);  // (B,H,L,HD)
  _Float16* Kf  = (_Float16*)(ws + 32 * MB);  // (B,H,L,HD)
  _Float16* VfT = (_Float16*)(ws + 40 * MB);  // (B,H,HD,L)  transposed
  _Float16* Tq  = (_Float16*)(ws + 48 * MB);  // (H,HD,V)
  _Float16* Tk  = (_Float16*)(ws + 50 * MB);
  _Float16* Tv  = (_Float16*)(ws + 52 * MB);
  float*    Hd  = (float*)   (ws + 54 * MB);  // (B,L,V)

  // 0) precision staging: activations f32->f16, weights transpose+convert
  cvt_kernel<<<dim3((Bc * Lc * Vc) / (256 * 8), 3), 256, 0, stream>>>(
      query, key_t, value, Xq, Xk, Xv);
  wt_kernel<<<dim3((Hc * HDc * Vc) / 256, 3), 256, 0, stream>>>(
      Wq, Wk, Wv, Tq, Tk, Tv);

  // 1) Q/K/V projections (V written transposed)
  proj_kernel<<<dim3(Lc / 64, 3 * Bc * Hc), 128, 0, stream>>>(
      Xq, Xk, Xv, Tq, Tk, Tv, Qf, Kf, VfT);

  // 2) per-(b,h) slab: scores -> softmax+masks -> attn*V   (S/At alias the X region,
  //    which is only read by proj_kernel; stream order makes this safe)
  for (int bh = 0; bh < Bc * Hc; ++bh) {
    int b = bh / Hc;
    int h = bh % Hc;
    const _Float16* Qs = Qf  + (size_t)bh * Lc * HDc;
    const _Float16* Ks = Kf  + (size_t)bh * Lc * HDc;
    const _Float16* Vs = VfT + (size_t)bh * HDc * Lc;
    float* Hs = Hd + (size_t)b * Lc * Vc + (size_t)h * HDc;

    scores_kernel<<<dim3(Lc / 16, 8), 128, 0, stream>>>(Qs, Ks, S);
    softmax_kernel<<<Lc, 256, 0, stream>>>(S, At, qmask + (size_t)b * Lc,
                                           vmask + (size_t)b * Lc);
    av_kernel<<<Lc / 64, 128, 0, stream>>>(At, Vs, Hs);
  }

  // 3) residual + LayerNorm
  ln_kernel<<<Bc * Lc, 256, 0, stream>>>(Hd, query, gamma, beta, out);
}
